// LFR_22471268893314
// MI455X (gfx1250) — compile-verified, matrix operand order checked by reading, not run on hardware
//
#include <hip/hip_runtime.h>
#include <hip/hip_bf16.h>
#include <stdint.h>

typedef __attribute__((ext_vector_type(16))) _Float16 v16h;
typedef __attribute__((ext_vector_type(8)))  _Float16 v8h;
typedef __attribute__((ext_vector_type(8)))  float    v8f;

#define BN_INV_F 0.99999500003749968f  // 1/sqrt(1+1e-5)

// ---------------------------------------------------------------------------
// Weight pre-swizzle into exact WMMA B-fragment layout.
// Bsw[((kc*NTILES + nt)*32 + lane)*16 + e] = W(k, n), with the ISA K striping:
//   lane<16 : e0..7 -> K = kc*32+e      ; e8..15 -> K = kc*32+16+(e-8)
//   lane>=16: e0..7 -> K = kc*32+8+e    ; e8..15 -> K = kc*32+24+(e-8)
// k ordering for 3x3: k = (ky*3+kx)*CIN + ci  (matches conv kernels below)
// ---------------------------------------------------------------------------
__global__ void k_swizzle(const float* __restrict__ w, _Float16* __restrict__ bsw,
                          int CIN, int KTOT, int NTOT, int NREAL, int is3x3)
{
  int idx = blockIdx.x * blockDim.x + threadIdx.x;
  if (idx >= KTOT * NTOT) return;
  int e    = idx & 15;
  int lane = (idx >> 4) & 31;
  int rest = idx >> 9;
  int NTILES = NTOT >> 4;
  int nt = rest % NTILES;
  int kc = rest / NTILES;
  int khalf = lane >> 4;
  int k = kc * 32 + khalf * 8 + (e < 8 ? e : e + 8);
  int n = nt * 16 + (lane & 15);
  float v = 0.0f;
  if (n < NREAL && k < KTOT) {
    if (is3x3) {
      int kidx = k / CIN, ci = k % CIN;
      int ky = kidx / 3, kx = kidx % 3;
      v = w[(((size_t)n * CIN + ci) * 3 + ky) * 3 + kx];
    } else {
      v = w[(size_t)n * CIN + k];
    }
  }
  bsw[idx] = (_Float16)v;
}

__device__ __forceinline__ v16h pack_afrag(v8h lo, v8h hi)
{
  v16h a;
#pragma unroll
  for (int i = 0; i < 8; ++i) { a[i] = lo[i]; a[i + 8] = hi[i]; }
  return a;
}

// ---------------------------------------------------------------------------
// f32 -> f16 conversions
// ---------------------------------------------------------------------------
__global__ void k_tof16(const float* __restrict__ in, _Float16* __restrict__ out, int n)
{
  int i = blockIdx.x * blockDim.x + threadIdx.x;
  if (i < n) out[i] = (_Float16)in[i];
}

// x_l (B,128*128,128 NHWC f32) -> cat2 channels 128..255 (stride 256)
__global__ void k_xl_to_cat2(const float* __restrict__ x_l, _Float16* __restrict__ cat2)
{
  int i = blockIdx.x * blockDim.x + threadIdx.x;
  if (i >= 65536 * 128) return;
  int pix = i >> 7, c = i & 127;
  cat2[(size_t)pix * 256 + 128 + c] = (_Float16)x_l[i];
}

// ---------------------------------------------------------------------------
// SE block: mean over HW, 128 ->relu 32 ->sigmoid 128, store gate as f16
// one block per batch (512 threads, LDS tree reduction)
// ---------------------------------------------------------------------------
__global__ __launch_bounds__(512)
void k_se(const float* __restrict__ x_l,
          const float* __restrict__ wf1, const float* __restrict__ bf1,
          const float* __restrict__ wf2, const float* __restrict__ bf2,
          _Float16* __restrict__ sgate)
{
  int b = blockIdx.x;
  int t = threadIdx.x;
  int c = t & 127, g = t >> 7;  // 4 pixel-groups x 128 channels
  const float* base = x_l + (size_t)b * 16384 * 128;
  float acc = 0.0f;
  for (int p = g; p < 16384; p += 4) acc += base[(size_t)p * 128 + c];
  __shared__ float lds[512];
  __shared__ float mean[128];
  __shared__ float g1[32];
  lds[t] = acc;
  __syncthreads();
  if (t < 128) mean[t] = (lds[t] + lds[t + 128] + lds[t + 256] + lds[t + 384]) * (1.0f / 16384.0f);
  __syncthreads();
  if (t < 32) {
    float a = bf1[t];
    for (int i = 0; i < 128; ++i) a += wf1[t * 128 + i] * mean[i];
    g1[t] = fmaxf(a, 0.0f);
  }
  __syncthreads();
  if (t < 128) {
    float a = bf2[t];
    for (int i = 0; i < 32; ++i) a += wf2[t * 32 + i] * g1[i];
    sgate[b * 128 + t] = (_Float16)(1.0f / (1.0f + __expf(-a)));
  }
}

// ---------------------------------------------------------------------------
// Generic 1x1-conv GEMM (WMMA): out = relu(acc*scale[n] + bias[n]) as f16
// one wave per block; MT m-tiles x NT n-tiles of 16x16
// ---------------------------------------------------------------------------
template<int MT, int NT>
__global__ __launch_bounds__(32)
void k_gemm_f16(const _Float16* __restrict__ A, int ASTRIDE, int KC,
                const _Float16* __restrict__ Bsw, int NTILES_TOT,
                const float* __restrict__ gma, const float* __restrict__ bet,
                _Float16* __restrict__ Out, int OSTRIDE, int OOFF)
{
  const int lane  = threadIdx.x;
  const int khalf = lane >> 4;
  const int l15   = lane & 15;
  const int m0 = blockIdx.x * (16 * MT);
  const int ng = blockIdx.y;
  v8f acc[MT][NT] = {};
  for (int cc = 0; cc < KC; ++cc) {
    v16h bf[NT];
#pragma unroll
    for (int t = 0; t < NT; ++t)
      bf[t] = *(const v16h*)(Bsw + (((size_t)cc * NTILES_TOT + (ng * NT + t)) * 32 + lane) * 16);
#pragma unroll
    for (int mt = 0; mt < MT; ++mt) {
      const _Float16* p = A + (size_t)(m0 + mt * 16 + l15) * ASTRIDE + cc * 32 + khalf * 8;
      v16h a = pack_afrag(*(const v8h*)p, *(const v8h*)(p + 16));
#pragma unroll
      for (int t = 0; t < NT; ++t)
        acc[mt][t] = __builtin_amdgcn_wmma_f32_16x16x32_f16(
            false, a, false, bf[t], (short)0, acc[mt][t], false, false);
    }
  }
#pragma unroll
  for (int mt = 0; mt < MT; ++mt) {
#pragma unroll
    for (int t = 0; t < NT; ++t) {
      const int n = (ng * NT + t) * 16 + l15;
      const float sc = gma[n] * BN_INV_F;
      const float bi = bet[n];
#pragma unroll
      for (int r = 0; r < 8; ++r) {
        const int m = m0 + mt * 16 + r + khalf * 8;
        float v = fmaxf(acc[mt][t][r] * sc + bi, 0.0f);
        Out[(size_t)m * OSTRIDE + OOFF + n] = (_Float16)v;
      }
    }
  }
}

// ---------------------------------------------------------------------------
// w2 conv: input = concat(sigmoid_gate * xl, xl), K=256, N=64 -> x12 ch 64..127
// xl lives at cat2[pix*256 + 128 .. 255]; gate is per (b, channel)
// ---------------------------------------------------------------------------
__global__ __launch_bounds__(32)
void k_conv_w2(const _Float16* __restrict__ cat2, const _Float16* __restrict__ sgate,
               const _Float16* __restrict__ Bsw,
               const float* __restrict__ gma, const float* __restrict__ bet,
               _Float16* __restrict__ x12)
{
  constexpr int MT = 2, NT = 4, NTILES_TOT = 4;
  const int lane  = threadIdx.x;
  const int khalf = lane >> 4;
  const int l15   = lane & 15;
  const int m0 = blockIdx.x * 32;
  const int b  = m0 >> 14;  // 16384 pixels per batch
  v8f acc[MT][NT] = {};
  for (int cc = 0; cc < 8; ++cc) {
    const bool gated = (cc < 4);
    const int ci = (gated ? cc : cc - 4) * 32 + khalf * 8;
    v16h bf[NT];
#pragma unroll
    for (int t = 0; t < NT; ++t)
      bf[t] = *(const v16h*)(Bsw + (((size_t)cc * NTILES_TOT + t) * 32 + lane) * 16);
    v8h sl = {}, sh = {};
    if (gated) {
      sl = *(const v8h*)(sgate + b * 128 + ci);
      sh = *(const v8h*)(sgate + b * 128 + ci + 16);
    }
#pragma unroll
    for (int mt = 0; mt < MT; ++mt) {
      const _Float16* p = cat2 + (size_t)(m0 + mt * 16 + l15) * 256 + 128 + ci;
      v8h lo = *(const v8h*)p;
      v8h hi = *(const v8h*)(p + 16);
      if (gated) { lo = lo * sl; hi = hi * sh; }
      v16h a = pack_afrag(lo, hi);
#pragma unroll
      for (int t = 0; t < NT; ++t)
        acc[mt][t] = __builtin_amdgcn_wmma_f32_16x16x32_f16(
            false, a, false, bf[t], (short)0, acc[mt][t], false, false);
    }
  }
#pragma unroll
  for (int mt = 0; mt < MT; ++mt) {
#pragma unroll
    for (int t = 0; t < NT; ++t) {
      const int n = t * 16 + l15;
      const float sc = gma[n] * BN_INV_F;
      const float bi = bet[n];
#pragma unroll
      for (int r = 0; r < 8; ++r) {
        const int m = m0 + mt * 16 + r + khalf * 8;
        float v = fmaxf(acc[mt][t][r] * sc + bi, 0.0f);
        x12[(size_t)m * 128 + 64 + n] = (_Float16)v;
      }
    }
  }
}

// ---------------------------------------------------------------------------
// 3x3 conv (pad=1) as WMMA GEMM with on-the-fly im2col.
// A: NHWC f16 with CIN channels. K ordering = (ky*3+kx)*CIN + ci.
// F16OUT: relu+f16 out; else raw f32 out (for `we` before softmax).
// ---------------------------------------------------------------------------
template<int CIN, int HH, int WW, int MT, int NT, bool F16OUT>
__global__ __launch_bounds__(32)
void k_conv3x3(const _Float16* __restrict__ A,
               const _Float16* __restrict__ Bsw, int NTILES_TOT,
               const float* __restrict__ gma, const float* __restrict__ bet, int NREAL,
               void* __restrict__ Out, int OSTRIDE, int OOFF)
{
  const int lane  = threadIdx.x;
  const int khalf = lane >> 4;
  const int l15   = lane & 15;
  const int ngroups = NTILES_TOT / NT;
  const int b  = blockIdx.z / ngroups;
  const int ng = blockIdx.z % ngroups;
  const int y  = blockIdx.y;
  const int x0 = blockIdx.x * (16 * MT);
  constexpr int KCC = CIN / 32;

  v8f acc[MT][NT] = {};
#pragma unroll
  for (int ky = 0; ky < 3; ++ky) {
    const int sy = y + ky - 1;
    const bool yok = (unsigned)sy < (unsigned)HH;
#pragma unroll
    for (int kx = 0; kx < 3; ++kx) {
      for (int cc = 0; cc < KCC; ++cc) {
        const int kcg = (ky * 3 + kx) * KCC + cc;
        v16h bf[NT];
#pragma unroll
        for (int t = 0; t < NT; ++t)
          bf[t] = *(const v16h*)(Bsw + (((size_t)kcg * NTILES_TOT + (ng * NT + t)) * 32 + lane) * 16);
#pragma unroll
        for (int mt = 0; mt < MT; ++mt) {
          const int sx = x0 + mt * 16 + l15 + kx - 1;
          v16h a = {};
          if (yok && (unsigned)sx < (unsigned)WW) {
            const _Float16* p = A + ((size_t)((b * HH + sy) * WW + sx)) * CIN + cc * 32 + khalf * 8;
            v8h lo = *(const v8h*)p;
            v8h hi = *(const v8h*)(p + 16);
#pragma unroll
            for (int i = 0; i < 8; ++i) { a[i] = lo[i]; a[i + 8] = hi[i]; }
          }
#pragma unroll
          for (int t = 0; t < NT; ++t)
            acc[mt][t] = __builtin_amdgcn_wmma_f32_16x16x32_f16(
                false, a, false, bf[t], (short)0, acc[mt][t], false, false);
        }
      }
    }
  }

#pragma unroll
  for (int mt = 0; mt < MT; ++mt) {
#pragma unroll
    for (int t = 0; t < NT; ++t) {
      const int n = (ng * NT + t) * 16 + l15;
      const float sc = (n < NREAL) ? gma[n] * BN_INV_F : 0.0f;
      const float bi = (n < NREAL) ? bet[n] : 0.0f;
#pragma unroll
      for (int r = 0; r < 8; ++r) {
        const int px = x0 + mt * 16 + r + khalf * 8;
        const size_t pix = (size_t)(b * HH + y) * WW + px;
        float v = acc[mt][t][r] * sc + bi;
        if (F16OUT) {
          v = fmaxf(v, 0.0f);
          ((_Float16*)Out)[pix * OSTRIDE + OOFF + n] = (_Float16)v;
        } else {
          ((float*)Out)[pix * OSTRIDE + OOFF + n] = v;
        }
      }
    }
  }
}

// ---------------------------------------------------------------------------
// wo2 1x1 conv: y1 (f16, NHWC stride 128) -> d_out f32 NCHW, bn+relu
// ---------------------------------------------------------------------------
__global__ __launch_bounds__(32)
void k_conv_wo2(const _Float16* __restrict__ y1, const _Float16* __restrict__ Bsw,
                const float* __restrict__ gma, const float* __restrict__ bet,
                float* __restrict__ out)
{
  constexpr int MT = 2, NT = 4, NTILES_TOT = 8;
  const int lane  = threadIdx.x;
  const int khalf = lane >> 4;
  const int l15   = lane & 15;
  const int m0 = blockIdx.x * 32;
  const int ng = blockIdx.y;
  v8f acc[MT][NT] = {};
  for (int cc = 0; cc < 4; ++cc) {
    v16h bf[NT];
#pragma unroll
    for (int t = 0; t < NT; ++t)
      bf[t] = *(const v16h*)(Bsw + (((size_t)cc * NTILES_TOT + (ng * NT + t)) * 32 + lane) * 16);
#pragma unroll
    for (int mt = 0; mt < MT; ++mt) {
      const _Float16* p = y1 + (size_t)(m0 + mt * 16 + l15) * 128 + cc * 32 + khalf * 8;
      v16h a = pack_afrag(*(const v8h*)p, *(const v8h*)(p + 16));
#pragma unroll
      for (int t = 0; t < NT; ++t)
        acc[mt][t] = __builtin_amdgcn_wmma_f32_16x16x32_f16(
            false, a, false, bf[t], (short)0, acc[mt][t], false, false);
    }
  }
#pragma unroll
  for (int mt = 0; mt < MT; ++mt) {
#pragma unroll
    for (int t = 0; t < NT; ++t) {
      const int n = (ng * NT + t) * 16 + l15;
      const float sc = gma[n] * BN_INV_F;
      const float bi = bet[n];
#pragma unroll
      for (int r = 0; r < 8; ++r) {
        const int m = m0 + mt * 16 + r + khalf * 8;
        const int b = m >> 14;
        const int rem = m & 16383;  // y*128+x
        float v = fmaxf(acc[mt][t][r] * sc + bi, 0.0f);
        out[((size_t)(b * 128 + n) << 14) + rem] = v;
      }
    }
  }
}

// ---------------------------------------------------------------------------
// Bilinear upsample 64x64 -> 128x128, align_corners (linspace endpoints)
// In: [b][64][64][CC] f16 ; Out: pixel stride `ostride`, channel offset `ooff`
// ---------------------------------------------------------------------------
template<int CC>
__global__ void k_upsample(const _Float16* __restrict__ In, _Float16* __restrict__ Out,
                           int ostride, int ooff)
{
  int t = blockIdx.x * blockDim.x + threadIdx.x;
  if (t >= 4 * 128 * 128 * CC) return;
  int c = t % CC;
  int r = t / CC;
  int ox = r & 127; r >>= 7;
  int oy = r & 127; int b = r >> 7;
  float fy = oy * (63.0f / 127.0f);
  float fx = ox * (63.0f / 127.0f);
  int y0 = (int)fy, x0 = (int)fx;
  float wy = fy - (float)y0, wx = fx - (float)x0;
  int y1 = (y0 < 63) ? y0 + 1 : 63;
  int x1 = (x0 < 63) ? x0 + 1 : 63;
  const _Float16* base = In + (size_t)b * 4096 * CC;
  float v00 = (float)base[(size_t)(y0 * 64 + x0) * CC + c];
  float v01 = (float)base[(size_t)(y0 * 64 + x1) * CC + c];
  float v10 = (float)base[(size_t)(y1 * 64 + x0) * CC + c];
  float v11 = (float)base[(size_t)(y1 * 64 + x1) * CC + c];
  float top = v00 + (v01 - v00) * wx;
  float bot = v10 + (v11 - v10) * wx;
  float v = top + (bot - top) * wy;
  Out[(size_t)((b * 16384) + oy * 128 + ox) * ostride + ooff + c] = (_Float16)v;
}

// ---------------------------------------------------------------------------
// softmax over 25 kernel channels (layout [pix][32], first 25 valid)
// ---------------------------------------------------------------------------
__global__ void k_softmax(float* __restrict__ Wk)
{
  int pix = blockIdx.x * blockDim.x + threadIdx.x;
  if (pix >= 65536) return;
  float* p = Wk + (size_t)pix * 32;
  float mx = p[0];
#pragma unroll
  for (int i = 1; i < 25; ++i) mx = fmaxf(mx, p[i]);
  float ex[25];
  float s = 0.0f;
#pragma unroll
  for (int i = 0; i < 25; ++i) { ex[i] = __expf(p[i] - mx); s += ex[i]; }
  float inv = 1.0f / s;
#pragma unroll
  for (int i = 0; i < 25; ++i) p[i] = ex[i] * inv;
}

// ---------------------------------------------------------------------------
// CARAFE reassembly: X[b,y,x,c] = sum_{i,j} Wk[pix][i*5+j]*xup[b,y+2i-4,x+2j-4,c]
// one block per pixel (128 threads = channels), weights staged in LDS
// writes X into cat2 channels 0..127
// ---------------------------------------------------------------------------
__global__ __launch_bounds__(128)
void k_carafe(const float* __restrict__ Wk, const _Float16* __restrict__ xup,
              _Float16* __restrict__ cat2)
{
  int pix = blockIdx.x;
  int c = threadIdx.x;
  int b = pix >> 14, rem = pix & 16383;
  int y = rem >> 7, x = rem & 127;
  __shared__ float wk[32];
  if (c < 32) wk[c] = Wk[(size_t)pix * 32 + c];
  __syncthreads();
  float acc = 0.0f;
#pragma unroll
  for (int i = 0; i < 5; ++i) {
    int sy = y + 2 * i - 4;
    if ((unsigned)sy >= 128u) continue;
#pragma unroll
    for (int j = 0; j < 5; ++j) {
      int sx = x + 2 * j - 4;
      if ((unsigned)sx >= 128u) continue;
      acc += wk[i * 5 + j] * (float)xup[((size_t)((b * 128 + sy) * 128) + sx) * 128 + c];
    }
  }
  cat2[(size_t)pix * 256 + c] = (_Float16)acc;
}

// ---------------------------------------------------------------------------
extern "C" void kernel_launch(void* const* d_in, const int* in_sizes, int n_in,
                              void* d_out, int out_size, void* d_ws, size_t ws_size,
                              hipStream_t stream)
{
  (void)in_sizes; (void)n_in; (void)out_size; (void)ws_size;
  const float* x_h   = (const float*)d_in[0];
  const float* x_l   = (const float*)d_in[1];
  const float* wr    = (const float*)d_in[2];
  const float* wr_g  = (const float*)d_in[3];
  const float* wr_b  = (const float*)d_in[4];
  const float* w1    = (const float*)d_in[5];
  const float* w1_g  = (const float*)d_in[6];
  const float* w1_b  = (const float*)d_in[7];
  const float* w2    = (const float*)d_in[8];
  const float* w2_g  = (const float*)d_in[9];
  const float* w2_b  = (const float*)d_in[10];
  const float* we    = (const float*)d_in[11];
  const float* we_g  = (const float*)d_in[12];
  const float* we_b  = (const float*)d_in[13];
  const float* wf1   = (const float*)d_in[14];
  const float* bf1   = (const float*)d_in[15];
  const float* wf2   = (const float*)d_in[16];
  const float* bf2   = (const float*)d_in[17];
  const float* wo1   = (const float*)d_in[18];
  const float* wo1_g = (const float*)d_in[19];
  const float* wo1_b = (const float*)d_in[20];
  const float* wo2   = (const float*)d_in[21];
  const float* wo2_g = (const float*)d_in[22];
  const float* wo2_b = (const float*)d_in[23];
  float* out = (float*)d_out;

  // workspace carve-up (256-byte aligned), ~104 MB total
  char* ws = (char*)d_ws;
  size_t off = 0;
  auto take = [&](size_t bytes) -> char* {
    char* p = ws + off;
    off += (bytes + 255) & ~(size_t)255;
    return p;
  };
  _Float16* xh   = (_Float16*)take(16384ull * 128 * 2);   // x_h as f16 NHWC (64x64x128)
  _Float16* cat2 = (_Float16*)take(65536ull * 256 * 2);   // [X | xl] 128x128x256
  _Float16* x12  = (_Float16*)take(65536ull * 128 * 2);   // [x1 | x2] 128x128x128
  _Float16* t1   = (_Float16*)take(16384ull * 64 * 2);    // conv(w1) pre-upsample 64x64x64
  _Float16* xr   = (_Float16*)take(16384ull * 128 * 2);   // conv(wr) 64x64x128
  _Float16* xup  = (_Float16*)take(65536ull * 128 * 2);   // upsampled xr 128x128x128
  _Float16* y1v  = (_Float16*)take(65536ull * 128 * 2);   // conv(wo1) out
  float*    Wk   = (float*)   take(65536ull * 32 * 4);    // kernel logits / softmax
  _Float16* sgat = (_Float16*)take(4ull * 128 * 2);       // SE gate
  _Float16* wrs  = (_Float16*)take(1152ull * 128 * 2);
  _Float16* w1s  = (_Float16*)take(128ull * 64 * 2);
  _Float16* w2s  = (_Float16*)take(256ull * 64 * 2);
  _Float16* wes  = (_Float16*)take(1152ull * 32 * 2);
  _Float16* wo1s = (_Float16*)take(2304ull * 128 * 2);
  _Float16* wo2s = (_Float16*)take(128ull * 128 * 2);

  auto g1d = [](long long n, int bs) { return dim3((unsigned)((n + bs - 1) / bs)); };

  // weight swizzles
  k_swizzle<<<g1d(1152 * 128, 256), 256, 0, stream>>>(wr,  wrs,  128, 1152, 128, 128, 1);
  k_swizzle<<<g1d(128 * 64,   256), 256, 0, stream>>>(w1,  w1s,  128, 128,  64,  64,  0);
  k_swizzle<<<g1d(256 * 64,   256), 256, 0, stream>>>(w2,  w2s,  256, 256,  64,  64,  0);
  k_swizzle<<<g1d(1152 * 32,  256), 256, 0, stream>>>(we,  wes,  128, 1152, 32,  25,  1);
  k_swizzle<<<g1d(2304 * 128, 256), 256, 0, stream>>>(wo1, wo1s, 256, 2304, 128, 128, 1);
  k_swizzle<<<g1d(128 * 128,  256), 256, 0, stream>>>(wo2, wo2s, 128, 128,  128, 128, 0);

  // activation prep
  k_tof16<<<g1d(16384 * 128, 256), 256, 0, stream>>>(x_h, xh, 16384 * 128);
  k_xl_to_cat2<<<g1d(65536LL * 128, 256), 256, 0, stream>>>(x_l, cat2);

  // SE gate
  k_se<<<dim3(4), 512, 0, stream>>>(x_l, wf1, bf1, wf2, bf2, sgat);

  // x1 = upsample(relu(bn(conv1x1(xh, w1))))  -> x12 ch 0..63
  k_gemm_f16<2, 4><<<dim3(512, 1), 32, 0, stream>>>(xh, 128, 4, w1s, 4, w1_g, w1_b, t1, 64, 0);
  k_upsample<64><<<g1d(4LL * 128 * 128 * 64, 256), 256, 0, stream>>>(t1, x12, 128, 0);

  // x2 = relu(bn(conv1x1(concat(gate*xl, xl), w2)))  -> x12 ch 64..127
  k_conv_w2<<<dim3(2048), 32, 0, stream>>>(cat2, sgat, w2s, w2_g, w2_b, x12);

  // Wk = softmax(bn(conv3x3(x12, we)))
  k_conv3x3<128, 128, 128, 2, 2, false><<<dim3(4, 128, 4), 32, 0, stream>>>(
      x12, wes, 2, we_g, we_b, 25, (void*)Wk, 32, 0);
  k_softmax<<<g1d(65536, 256), 256, 0, stream>>>(Wk);

  // xr = relu(bn(conv3x3(xh, wr))); xup = upsample(xr)
  k_conv3x3<128, 64, 64, 2, 4, true><<<dim3(2, 64, 8), 32, 0, stream>>>(
      xh, wrs, 8, wr_g, wr_b, 128, (void*)xr, 128, 0);
  k_upsample<128><<<g1d(4LL * 128 * 128 * 128, 256), 256, 0, stream>>>(xr, xup, 128, 0);

  // CARAFE reassembly -> cat2 ch 0..127
  k_carafe<<<dim3(65536), 128, 0, stream>>>(Wk, xup, cat2);

  // y1 = relu(bn(conv3x3(cat2, wo1)))
  k_conv3x3<256, 128, 128, 2, 4, true><<<dim3(4, 128, 8), 32, 0, stream>>>(
      cat2, wo1s, 8, wo1_g, wo1_b, 128, (void*)y1v, 128, 0);

  // out = relu(bn(conv1x1(y1, wo2)))  -> NCHW f32
  k_conv_wo2<<<dim3(2048, 2), 32, 0, stream>>>(y1v, wo2s, wo2_g, wo2_b, out);
}
